// RGCN_23089744183803
// MI455X (gfx1250) — compile-verified, matrix operand order checked by reading, not run on hardware
//
#include <hip/hip_runtime.h>
#include <stdint.h>

// ---------------------------------------------------------------------------
// RGCN (2-layer HeteroGraphConv, norm='both') for MI455X / gfx1250 (wave32).
//
// Pipeline (all linear steps commuted to transform-first + gather-by-dst):
//   1. integer degree counts (atomics)           -> out_norm/in_norm = deg^-1/2
//   2. single-block exclusive scan of in-degrees -> CSR row offsets
//   3. bucket fill (atomic cursor)               -> csr_src[], dst-sorted src ids
//   4. z_r = (out_norm_r * x) @ W1_r   (WMMA f16->f32, per relation)
//   5. h[d] = relu( sum_r in_norm_r[d] * sum_{e: dst=d} z_r[src_e] + sum_r b1_r )
//   6. z2_r = (out_norm_r * h) @ W2_r  (WMMA)
//   7. out[d] = sum_r in_norm_r[d] * sum z2_r[src_e] + sum_r b2_r
// ---------------------------------------------------------------------------

typedef _Float16 half16 __attribute__((ext_vector_type(16)));
typedef float    float8 __attribute__((ext_vector_type(8)));

static constexpr int KPAD = 72;   // 64-wide K stage + 8-half pad (breaks 256B bank period)

// ------------------------- trivial utility kernels -------------------------

__global__ void zero_u32_kernel(unsigned* __restrict__ p, int n) {
    int i = blockIdx.x * blockDim.x + threadIdx.x;
    if (i < n) p[i] = 0u;
}

__global__ void count_deg_kernel(const int* __restrict__ src, const int* __restrict__ dst,
                                 unsigned* __restrict__ dout, unsigned* __restrict__ din,
                                 int nodes, int epr, int total) {
    int i = blockIdx.x * blockDim.x + threadIdx.x;
    if (i >= total) return;
    int r = i / epr;
    atomicAdd(&dout[(size_t)r * nodes + src[i]], 1u);
    atomicAdd(&din [(size_t)r * nodes + dst[i]], 1u);
}

__global__ void norm_kernel(const unsigned* __restrict__ dout, const unsigned* __restrict__ din,
                            float* __restrict__ onorm, float* __restrict__ inorm, int n) {
    int i = blockIdx.x * blockDim.x + threadIdx.x;
    if (i >= n) return;
    unsigned o = dout[i]; if (o == 0u) o = 1u;
    unsigned d = din[i];  if (d == 0u) d = 1u;
    onorm[i] = 1.0f / sqrtf((float)o);
    inorm[i] = 1.0f / sqrtf((float)d);
}

// Single-block exclusive scan over n (=R*N=150000) uints. 1024 threads,
// 4 elems/thread, shfl-based wave scans (wave32) -> ~37 chunk iterations.
__global__ void scan_kernel(const unsigned* __restrict__ deg,
                            unsigned* __restrict__ row_start,
                            unsigned* __restrict__ cursor, int n) {
    __shared__ unsigned waveSums[32];
    __shared__ unsigned waveOff[32];
    __shared__ unsigned sCarry;
    __shared__ unsigned sTot;
    const int tid  = threadIdx.x;
    const int lane = tid & 31;
    const int wid  = tid >> 5;
    if (tid == 0) sCarry = 0u;
    __syncthreads();
    const int CHUNK = 1024 * 4;
    for (int base = 0; base < n; base += CHUNK) {
        int i0 = base + tid * 4;
        unsigned a0 = 0, a1 = 0, a2 = 0, a3 = 0;
        if (i0 + 3 < n) {
            uint4 v = *(const uint4*)(deg + i0);
            a0 = v.x; a1 = v.y; a2 = v.z; a3 = v.w;
        } else {
            if (i0     < n) a0 = deg[i0];
            if (i0 + 1 < n) a1 = deg[i0 + 1];
            if (i0 + 2 < n) a2 = deg[i0 + 2];
            if (i0 + 3 < n) a3 = deg[i0 + 3];
        }
        unsigned p0 = a0, p1 = p0 + a1, p2 = p1 + a2, p3 = p2 + a3;
        unsigned incl = p3;
#pragma unroll
        for (int d = 1; d < 32; d <<= 1) {
            unsigned t = __shfl_up(incl, (unsigned)d, 32);
            if (lane >= d) incl += t;
        }
        unsigned wexcl = incl - p3;
        if (lane == 31) waveSums[wid] = incl;
        __syncthreads();
        if (wid == 0) {
            unsigned v = waveSums[lane];
            unsigned inc2 = v;
#pragma unroll
            for (int d = 1; d < 32; d <<= 1) {
                unsigned t = __shfl_up(inc2, (unsigned)d, 32);
                if (lane >= d) inc2 += t;
            }
            waveOff[lane] = inc2 - v;
            if (lane == 31) sTot = inc2;
        }
        __syncthreads();
        unsigned bse = sCarry + waveOff[wid] + wexcl;
        if (i0     < n) { row_start[i0]     = bse;      cursor[i0]     = bse;      }
        if (i0 + 1 < n) { row_start[i0 + 1] = bse + p0; cursor[i0 + 1] = bse + p0; }
        if (i0 + 2 < n) { row_start[i0 + 2] = bse + p1; cursor[i0 + 2] = bse + p1; }
        if (i0 + 3 < n) { row_start[i0 + 3] = bse + p2; cursor[i0 + 3] = bse + p2; }
        __syncthreads();
        if (tid == 0) sCarry += sTot;
        __syncthreads();
    }
    if (threadIdx.x == 0) row_start[n] = sCarry;
}

__global__ void fill_csr_kernel(const int* __restrict__ src, const int* __restrict__ dst,
                                unsigned* __restrict__ cursor, unsigned* __restrict__ csr_src,
                                int nodes, int epr, int total) {
    int i = blockIdx.x * blockDim.x + threadIdx.x;
    if (i >= total) return;
    int r = i / epr;
    unsigned b = (unsigned)r * (unsigned)nodes + (unsigned)dst[i];
    unsigned pos = atomicAdd(&cursor[b], 1u);
    csr_src[pos] = (unsigned)src[i];
}

// ------------------------------ WMMA GEMM ----------------------------------
// Fragment assembly for the documented wave32 16-bit A/B layouts:
// lane(0-15)=row/col, K packed 2/VGPR; VGPR0-3: K 0..7 (+8 for lanes 16-31),
// VGPR4-7: K 16..23 (+8). `p` points at a 16-row tile with row stride KPAD.
__device__ inline half16 load_frag(const _Float16* p, int kb, int lane) {
    const int r  = lane & 15;
    const int hi = lane >> 4;
    const uint32_t* rp = reinterpret_cast<const uint32_t*>(p + r * KPAD + kb * 32 + hi * 8);
    union { uint32_t u[8]; half16 v; } t;
#pragma unroll
    for (int i = 0; i < 8; ++i)
        t.u[i] = rp[(i & 3) + ((i >> 2) << 3)];
    return t.v;
}

// Z_r = (norm_r ⊙ X) @ W_r.  X:[nrows,128] f32, W:[R,128,OUTW] f32 row-major.
// Block = 256 threads (8 waves), 128-row panel, K in two 64-wide stages with
// an explicit regs->LDS pipeline: stage-2 global loads are issued *before*
// stage-1 compute so their latency hides under 16 WMMAs.  grid.y = relation.
template <int OUTW>
__global__ __launch_bounds__(256) void gemm_kernel(const float* __restrict__ X,
                                                   const float* __restrict__ norms,
                                                   const float* __restrict__ Wall,
                                                   float* __restrict__ Zall,
                                                   int nrows) {
    __shared__ _Float16 sA[128 * KPAD];
    __shared__ _Float16 sB[OUTW * KPAD];
    const int tid  = threadIdx.x;
    const int lane = tid & 31;
    const int wid  = tid >> 5;
    const int rel  = blockIdx.y;
    const int row0 = blockIdx.x * 128;
    const float* nrm = norms + (size_t)rel * nrows;
    const float* W   = Wall  + (size_t)rel * (128 * OUTW);
    float*       Z   = Zall  + (size_t)rel * nrows * OUTW;

    constexpr int NT  = OUTW / 16;    // output tiles per wave
    constexpr int BF4 = OUTW / 16;    // float4 loads per thread for the B stage

    float8 acc[NT];
#pragma unroll
    for (int nt = 0; nt < NT; ++nt)
#pragma unroll
        for (int j = 0; j < 8; ++j) acc[nt][j] = 0.0f;

    float4 ga[8];  float gnA[8];      // A stage regs: 128 rows x 64 K / 256 thr
    float4 gb[BF4];                   // B stage regs: 64 K x OUTW / 256 thr

    auto loadA = [&](int k0) {
#pragma unroll
        for (int i = 0; i < 8; ++i) {
            int idx = i * 256 + tid;
            int rr  = idx >> 4;
            int c4  = idx & 15;
            int gr  = row0 + rr;
            if (gr < nrows) {
                ga[i]  = *(const float4*)(X + (size_t)gr * 128 + k0 + c4 * 4);
                gnA[i] = nrm[gr];
            } else {
                ga[i]  = make_float4(0.f, 0.f, 0.f, 0.f);
                gnA[i] = 0.f;
            }
        }
    };
    auto storeA = [&]() {
#pragma unroll
        for (int i = 0; i < 8; ++i) {
            int idx = i * 256 + tid;
            int rr  = idx >> 4;
            int c4  = idx & 15;
            _Float16* dp = sA + rr * KPAD + c4 * 4;
            dp[0] = (_Float16)(ga[i].x * gnA[i]);
            dp[1] = (_Float16)(ga[i].y * gnA[i]);
            dp[2] = (_Float16)(ga[i].z * gnA[i]);
            dp[3] = (_Float16)(ga[i].w * gnA[i]);
        }
    };
    auto loadB = [&](int k0) {        // coalesced float4 along out-dim
#pragma unroll
        for (int i = 0; i < BF4; ++i) {
            int idx = i * 256 + tid;
            int kk  = idx / (OUTW / 4);
            int oo4 = idx % (OUTW / 4);
            gb[i] = *(const float4*)(W + (size_t)(k0 + kk) * OUTW + oo4 * 4);
        }
    };
    auto storeB = [&]() {             // transpose into sB[out][k] (f16)
#pragma unroll
        for (int i = 0; i < BF4; ++i) {
            int idx = i * 256 + tid;
            int kk  = idx / (OUTW / 4);
            int oo4 = idx % (OUTW / 4);
            _Float16* dp = sB + (oo4 * 4) * KPAD + kk;
            dp[0 * KPAD] = (_Float16)gb[i].x;
            dp[1 * KPAD] = (_Float16)gb[i].y;
            dp[2 * KPAD] = (_Float16)gb[i].z;
            dp[3 * KPAD] = (_Float16)gb[i].w;
        }
    };
    auto compute = [&]() {
#pragma unroll
        for (int kb = 0; kb < 2; ++kb) {
            half16 a    = load_frag(sA + wid * 16 * KPAD, kb, lane);
            half16 bcur = load_frag(sB, kb, lane);
#pragma unroll
            for (int nt = 0; nt < NT; ++nt) {
                // software pipeline: issue next B-fragment loads before WMMA
                half16 bnxt = (nt + 1 < NT) ? load_frag(sB + (nt + 1) * 16 * KPAD, kb, lane)
                                            : bcur;
                acc[nt] = __builtin_amdgcn_wmma_f32_16x16x32_f16(
                    false, a, false, bnxt == bnxt ? bcur : bcur, (short)0, acc[nt], false, false);
                bcur = bnxt;
            }
        }
    };

    // ---- explicit 2-stage pipeline over K ----
    loadA(0);  loadB(0);
    storeA();  storeB();
    __syncthreads();
    loadA(64); loadB(64);             // global latency overlaps stage-1 compute
    compute();
    __syncthreads();                  // all waves done reading stage-1 LDS
    storeA();  storeB();
    __syncthreads();
    compute();

    // C/D layout: VGPR v, lanes 0-15 -> M=v, lanes 16-31 -> M=v+8; N=lane&15.
    const int mBase = (lane >> 4) * 8;
    const int nCol  = lane & 15;
#pragma unroll
    for (int nt = 0; nt < NT; ++nt) {
#pragma unroll
        for (int v = 0; v < 8; ++v) {
            int gr = row0 + wid * 16 + mBase + v;
            if (gr < nrows) Z[(size_t)gr * OUTW + nt * 16 + nCol] = acc[nt][v];
        }
    }
}

// --------------------------- gather (by dst) -------------------------------
// One wave32 per destination node; lanes cover the feature dim with float4/2.
// Edge IDs for a node are loaded 32-at-a-time by the lanes and broadcast via
// shfl, removing the per-edge dependent csr load from the latency chain.
__global__ void gather1_kernel(const float* __restrict__ z,          // [3][N][128]
                               const unsigned* __restrict__ row_start,
                               const unsigned* __restrict__ csr_src,
                               const float* __restrict__ in_norm,    // [3*N]
                               const float* __restrict__ b1,         // [3][128]
                               float* __restrict__ h, int nodes) {
    int gw   = (blockIdx.x * blockDim.x + threadIdx.x) >> 5;
    int lane = threadIdx.x & 31;
    if (gw >= nodes) return;
    float a0 = 0.f, a1 = 0.f, a2 = 0.f, a3 = 0.f;
#pragma unroll
    for (int r = 0; r < 3; ++r) {
        unsigned b = (unsigned)r * (unsigned)nodes + (unsigned)gw;
        unsigned s = row_start[b], e = row_start[b + 1];
        float s0 = 0.f, s1 = 0.f, s2 = 0.f, s3 = 0.f;
        for (unsigned j0 = s; j0 < e; j0 += 32) {
            unsigned myid = (j0 + lane < e) ? csr_src[j0 + lane] : 0u;
            int cnt = (int)((e - j0 < 32u) ? (e - j0) : 32u);
            for (int t = 0; t < cnt; ++t) {
                unsigned sid = __shfl(myid, t, 32);
                float4 v = *(const float4*)(z + ((size_t)r * nodes + sid) * 128 + lane * 4);
                s0 += v.x; s1 += v.y; s2 += v.z; s3 += v.w;
            }
        }
        float inn = in_norm[b];
        a0 += inn * s0; a1 += inn * s1; a2 += inn * s2; a3 += inn * s3;
    }
    int c = lane * 4;
    float4 o;
    o.x = fmaxf(a0 + b1[c    ] + b1[128 + c    ] + b1[256 + c    ], 0.f);
    o.y = fmaxf(a1 + b1[c + 1] + b1[128 + c + 1] + b1[256 + c + 1], 0.f);
    o.z = fmaxf(a2 + b1[c + 2] + b1[128 + c + 2] + b1[256 + c + 2], 0.f);
    o.w = fmaxf(a3 + b1[c + 3] + b1[128 + c + 3] + b1[256 + c + 3], 0.f);
    *(float4*)(h + (size_t)gw * 128 + c) = o;
}

// Layer 2: width 64, fused bias-sum, writes d_out directly.
__global__ void gather2_kernel(const float* __restrict__ z,          // [3][N][64]
                               const unsigned* __restrict__ row_start,
                               const unsigned* __restrict__ csr_src,
                               const float* __restrict__ in_norm,
                               const float* __restrict__ b2,         // [3][64]
                               float* __restrict__ out, int nodes) {
    int gw   = (blockIdx.x * blockDim.x + threadIdx.x) >> 5;
    int lane = threadIdx.x & 31;
    if (gw >= nodes) return;
    float a0 = 0.f, a1 = 0.f;
#pragma unroll
    for (int r = 0; r < 3; ++r) {
        unsigned b = (unsigned)r * (unsigned)nodes + (unsigned)gw;
        unsigned s = row_start[b], e = row_start[b + 1];
        float s0 = 0.f, s1 = 0.f;
        for (unsigned j0 = s; j0 < e; j0 += 32) {
            unsigned myid = (j0 + lane < e) ? csr_src[j0 + lane] : 0u;
            int cnt = (int)((e - j0 < 32u) ? (e - j0) : 32u);
            for (int t = 0; t < cnt; ++t) {
                unsigned sid = __shfl(myid, t, 32);
                float2 v = *(const float2*)(z + ((size_t)r * nodes + sid) * 64 + lane * 2);
                s0 += v.x; s1 += v.y;
            }
        }
        float inn = in_norm[b];
        a0 += inn * s0; a1 += inn * s1;
    }
    int c = lane * 2;
    float2 o;
    o.x = a0 + b2[c    ] + b2[64 + c    ] + b2[128 + c    ];
    o.y = a1 + b2[c + 1] + b2[64 + c + 1] + b2[128 + c + 1];
    *(float2*)(out + (size_t)gw * 64 + c) = o;
}

// ------------------------------- launcher ----------------------------------

extern "C" void kernel_launch(void* const* d_in, const int* in_sizes, int n_in,
                              void* d_out, int out_size, void* d_ws, size_t ws_size,
                              hipStream_t stream) {
    const float* x  = (const float*)d_in[0];
    const float* W1 = (const float*)d_in[1];
    const float* b1 = (const float*)d_in[2];
    const float* W2 = (const float*)d_in[3];
    const float* b2 = (const float*)d_in[4];
    const int*  src = (const int*)d_in[5];
    const int*  dst = (const int*)d_in[6];

    const int N  = in_sizes[0] / 128;   // 50000
    const int RE = in_sizes[5];         // 3 * 200000
    const int E  = RE / 3;
    const int RN = 3 * N;

    // workspace carve-out (256B aligned slices)
    char* ws = (char*)d_ws;
    size_t off = 0;
    auto take = [&](size_t bytes) -> char* {
        char* p = ws + off;
        off += (bytes + 255) & ~(size_t)255;
        return p;
    };
    unsigned* degs      = (unsigned*)take((size_t)2 * RN * 4);   // out | in, contiguous
    unsigned* deg_out   = degs;
    unsigned* deg_in    = degs + RN;
    float*    out_norm  = (float*)take((size_t)RN * 4);
    float*    in_norm   = (float*)take((size_t)RN * 4);
    unsigned* row_start = (unsigned*)take((size_t)(RN + 1) * 4);
    unsigned* cursor    = (unsigned*)take((size_t)RN * 4);
    unsigned* csr_src   = (unsigned*)take((size_t)RE * 4);
    float*    z         = (float*)take((size_t)3 * N * 128 * 4); // reused for layer-2 [3][N][64]
    float*    h         = (float*)take((size_t)N * 128 * 4);
    (void)ws_size; (void)n_in; (void)out_size;

    // 1-3: degrees, norms, CSR
    zero_u32_kernel<<<(2 * RN + 255) / 256, 256, 0, stream>>>(degs, 2 * RN);
    count_deg_kernel<<<(RE + 255) / 256, 256, 0, stream>>>(src, dst, deg_out, deg_in, N, E, RE);
    norm_kernel<<<(RN + 255) / 256, 256, 0, stream>>>(deg_out, deg_in, out_norm, in_norm, RN);
    scan_kernel<<<1, 1024, 0, stream>>>(deg_in, row_start, cursor, RN);
    fill_csr_kernel<<<(RE + 255) / 256, 256, 0, stream>>>(src, dst, cursor, csr_src, N, E, RE);

    // layer 1
    dim3 ggrid((N + 127) / 128, 3);
    gemm_kernel<128><<<ggrid, 256, 0, stream>>>(x, out_norm, W1, z, N);
    gather1_kernel<<<(N * 32 + 255) / 256, 256, 0, stream>>>(z, row_start, csr_src, in_norm, b1, h, N);

    // layer 2
    gemm_kernel<64><<<ggrid, 256, 0, stream>>>(h, out_norm, W2, z, N);
    gather2_kernel<<<(N * 32 + 255) / 256, 256, 0, stream>>>(z, row_start, csr_src, in_norm, b2,
                                                             (float*)d_out, N);
}
